// MHGraphAttention1_6073083757191
// MI455X (gfx1250) — matrix operand order repaired and checked by reading, not compiled, not run on hardware
//
#include <hip/hip_runtime.h>
#include <math.h>

#define DIN    512
#define DKd    64
#define DVd    64
#define DRd    16
#define NHh    8
#define DMODEL 512
#define NSEQ   4096
#define TOTW   208            // DK + DK + DV + DR
#define NPROJ  (NHh * TOTW)   // 1664

typedef __attribute__((ext_vector_type(16))) __bf16 v16bf;
typedef __attribute__((ext_vector_type(8)))  float  v8f;
typedef __attribute__((ext_vector_type(4)))  unsigned int v4u;
typedef __attribute__((ext_vector_type(8)))  int    v8i;
typedef __attribute__((ext_vector_type(4)))  int    v4i;

union FragBF { v16bf v; uint4 u[2]; };

// round-to-nearest-even float -> bf16 bits (avoids scalar __bf16 arithmetic)
static __device__ __forceinline__ unsigned short f2bf(float x) {
    union { float f; unsigned int u; } t; t.f = x;
    unsigned int u = t.u;
    unsigned int lsb = (u >> 16) & 1u;
    u += 0x7fffu + lsb;
    return (unsigned short)(u >> 16);
}

// ---------------------------------------------------------------------------
// TDM: DMA a 2-D fp32 tile (tile_rows x 512 elements, row stride N*DR elems)
// from global memory into LDS.  Descriptor per CDNA5 ISA 8.3/8.4.
static __device__ __forceinline__ void tdm_load_d_tile(unsigned lds_off,
                                                       const float* gptr) {
    unsigned long long ga = (unsigned long long)(size_t)gptr;
    v4u g0 = { 1u,                                   // count=1, is_restore=0
               lds_off,                              // lds_addr
               (unsigned)ga,                         // global_addr[31:0]
               (unsigned)(ga >> 32) | (2u << 30) };  // global_addr[56:32] | type=2
    v8i g1 = { (int)0x00020000u,   // wg_mask=0, data_size=2 (4 bytes)
               (int)0x02000000u,   // tensor_dim0[15:0]=512  (<<16)
               (int)0x00100000u,   // tensor_dim1[15:0]=16   (<<16)
               (int)0x02000000u,   // tile_dim0=512          (<<16)
               (int)0x00000010u,   // tile_dim1=16, tile_dim2=0
               (int)0x00010000u,   // tensor_dim0_stride = 65536 (N*DR)
               0, 0 };             // dim0_stride hi / dim1_stride = 0
    v4i z4 = { 0, 0, 0, 0 };
#if __clang_major__ >= 23
    v8i z8 = { 0, 0, 0, 0, 0, 0, 0, 0 };
    __builtin_amdgcn_tensor_load_to_lds(g0, g1, z4, z4, z8, 0);
#else
    __builtin_amdgcn_tensor_load_to_lds(g0, g1, z4, z4, 0);
#endif
}

// ---------------------------------------------------------------------------
// elementwise fp32 -> bf16
__global__ void cvt_bf16_kernel(const float* __restrict__ src,
                                unsigned short* __restrict__ dst, int n) {
    int i = blockIdx.x * blockDim.x + threadIdx.x;
    int stride = gridDim.x * blockDim.x;
    for (; i < n; i += stride) dst[i] = f2bf(src[i]);
}

// transpose + convert: src[rows][cols] fp32 -> dst[cols][rows] bf16
__global__ void cvt_tr_bf16_kernel(const float* __restrict__ src,
                                   unsigned short* __restrict__ dst,
                                   int rows, int cols) {
    int i = blockIdx.x * blockDim.x + threadIdx.x;
    int n = rows * cols;
    int stride = gridDim.x * blockDim.x;
    for (; i < n; i += stride) {
        int r = i / cols, c = i - r * cols;
        dst[(size_t)c * rows + r] = f2bf(src[i]);
    }
}

// ---------------------------------------------------------------------------
// C[M][Nc] fp32 = A[M][K] bf16 (row major) * Bt[Nc][K] bf16 (B transposed)
// One wave per 16x16 C tile. K must be a multiple of 32.
__global__ __launch_bounds__(32)
void gemm_bf16_kernel(const unsigned short* __restrict__ A,
                      const unsigned short* __restrict__ Bt,
                      float* __restrict__ C, int M, int Nc, int K) {
    const int lane = threadIdx.x & 31;
    const int hi   = lane >> 4;       // lane half selects K sub-range
    const int n    = lane & 15;       // M (for A) / N (for B) index
    const int m0   = blockIdx.y * 16;
    const int n0   = blockIdx.x * 16;
    (void)M;

    const unsigned short* arow = A  + (size_t)(m0 + n) * K;
    const unsigned short* brow = Bt + (size_t)(n0 + n) * K;

    v8f c = {};
    for (int kk = 0; kk < K; kk += 32) {
        FragBF a, b;
        a.u[0] = *(const uint4*)(arow + kk + 8 * hi);        // K = 8hi..8hi+7
        a.u[1] = *(const uint4*)(arow + kk + 16 + 8 * hi);   // K = 16+8hi..+7
        b.u[0] = *(const uint4*)(brow + kk + 8 * hi);        // K = 8hi..8hi+7
        b.u[1] = *(const uint4*)(brow + kk + 16 + 8 * hi);   // K = 16+8hi..+7
        c = __builtin_amdgcn_wmma_f32_16x16x32_bf16(false, a.v, false, b.v,
                                                    (short)0, c, false, false);
    }
#pragma unroll
    for (int k = 0; k < 8; ++k)       // C layout: M = k + 8*hi, N = lane&15
        C[(size_t)(m0 + k + 8 * hi) * Nc + (n0 + n)] = c[k];
}

// ---------------------------------------------------------------------------
// Split H2[N][NPROJ] into per-head Q,K (bf16 row major), V transposed (bf16),
// R (fp32).
__global__ void reformat_kernel(const float* __restrict__ H2,
                                unsigned short* __restrict__ Qb,
                                unsigned short* __restrict__ Kb,
                                unsigned short* __restrict__ Vt,
                                float* __restrict__ Rf) {
    int i = blockIdx.x;
    int t = threadIdx.x;
    if (t >= TOTW) return;
    const float* row = H2 + (size_t)i * NPROJ;
#pragma unroll
    for (int h = 0; h < NHh; ++h) {
        float v = row[h * TOTW + t];
        if (t < DKd)
            Qb[((size_t)h * NSEQ + i) * DKd + t] = f2bf(v);
        else if (t < 2 * DKd)
            Kb[((size_t)h * NSEQ + i) * DKd + (t - DKd)] = f2bf(v);
        else if (t < 2 * DKd + DVd)
            Vt[((size_t)h * DVd + (t - 2 * DKd)) * NSEQ + i] = f2bf(v);
        else
            Rf[((size_t)h * NSEQ + i) * DRd + (t - 2 * DKd - DVd)] = v;
    }
}

// ---------------------------------------------------------------------------
// Fused flash attention: block = 16 query rows x ALL 8 heads (wave == head).
// The 1 GiB D tensor is streamed from HBM exactly once, via double-buffered
// TDM (tensor_load_to_lds) DMA into LDS, overlapped with the QK^T WMMAs.
//
// Dynamic LDS layout (81920 bytes):
//   [0,      65536)  float  ldsD[2][16][32*16]   two 32 KB D tiles
//   [65536,  73728)  ushort ldsP[8][16][32]      per-wave P staging
//   [73728,  81920)  float  ldsR[8][16][16]      per-head R tile
#define ATTN_LDS_BYTES 81920
#define LDSD_TILE_F    (16 * 32 * 16)   // floats per D buffer

__global__ __launch_bounds__(256)
void attn_kernel(const unsigned short* __restrict__ Qb,
                 const unsigned short* __restrict__ Kb,
                 const unsigned short* __restrict__ Vt,
                 const float* __restrict__ Rf,
                 const float* __restrict__ Dp,
                 const float* __restrict__ mask,
                 float* __restrict__ ctx) {
    extern __shared__ char smem[];
    float*          ldsD = (float*)smem;
    unsigned short* ldsP = (unsigned short*)(smem + 65536);
    float*          ldsR = (float*)(smem + 73728);

    const int w    = threadIdx.x >> 5;   // wave index == head
    const int lane = threadIdx.x & 31;
    const int hi   = lane >> 4;
    const int n    = lane & 15;
    const int h    = w;
    const int i0   = blockIdx.x * 16;
    const float scale = 0.11180339887498949f;      // 1/sqrt(DK+DR) = 1/sqrt(80)

    // stage this head's R tile (16 x 16 fp32) into LDS once
    for (int t = lane; t < 256; t += 32) {
        int row = t >> 4, col = t & 15;
        ldsR[(w * 16 + row) * 16 + col] =
            Rf[((size_t)h * NSEQ + i0 + row) * DRd + col];
    }

    // kick off the first D-tile DMA (wave 0 only; TDM ignores EXEC)
    if (w == 0)
        tdm_load_d_tile((unsigned)(size_t)(void*)smem,
                        Dp + ((size_t)i0 * NSEQ + 0) * DRd);

    // preload Q A-fragments (16 x 64 bf16) for this head's query tile
    const unsigned short* qrow = Qb + ((size_t)h * NSEQ + i0 + n) * DKd;
    FragBF qa[2];
#pragma unroll
    for (int c = 0; c < 2; ++c) {
        qa[c].u[0] = *(const uint4*)(qrow + 32 * c + 8 * hi);
        qa[c].u[1] = *(const uint4*)(qrow + 32 * c + 16 + 8 * hi);
    }

    v8f acc[4] = {};                    // ctx accumulators: 4 tiles of 16 d
    float m_r[8], l_r[8];
#pragma unroll
    for (int k = 0; k < 8; ++k) { m_r[k] = -__builtin_inff(); l_r[k] = 0.f; }

    int buf = 0;
    for (int j0 = 0; j0 < NSEQ; j0 += 32) {
        const bool has_next = (j0 + 32) < NSEQ;
        // prefetch next D tile into the other buffer
        if (w == 0 && has_next)
            tdm_load_d_tile((unsigned)(size_t)(void*)(smem + (buf ^ 1) * 32768),
                            Dp + ((size_t)i0 * NSEQ + (j0 + 32)) * DRd);

        // ---- scores: Q @ K^T for two 16-wide j sub-tiles (overlaps DMA) ----
        v8f s[2];
#pragma unroll
        for (int t = 0; t < 2; ++t) {
            const unsigned short* krow =
                Kb + ((size_t)h * NSEQ + j0 + 16 * t + n) * DKd;
            FragBF b0, b1;
            b0.u[0] = *(const uint4*)(krow + 8 * hi);
            b0.u[1] = *(const uint4*)(krow + 16 + 8 * hi);
            b1.u[0] = *(const uint4*)(krow + 32 + 8 * hi);
            b1.u[1] = *(const uint4*)(krow + 32 + 16 + 8 * hi);
            v8f z = {};
            z = __builtin_amdgcn_wmma_f32_16x16x32_bf16(false, qa[0].v, false,
                                                        b0.v, (short)0, z,
                                                        false, false);
            z = __builtin_amdgcn_wmma_f32_16x16x32_bf16(false, qa[1].v, false,
                                                        b1.v, (short)0, z,
                                                        false, false);
            s[t] = z;
        }

        // current D tile must have landed; barrier also fences last P reads
        if (w == 0) {
            if (has_next) __builtin_amdgcn_s_wait_tensorcnt(1);
            else          __builtin_amdgcn_s_wait_tensorcnt(0);
        }
        __syncthreads();

        // ---- RD[i,j] = dot(R[h,i,:], D[i,j,:]) from LDS, + mask, scale ----
        const float* Db = ldsD + buf * LDSD_TILE_F;
#pragma unroll
        for (int k = 0; k < 8; ++k) {
            int il = k + 8 * hi;                      // C layout: M = k + 8*hi
            const float4* Rp = (const float4*)(ldsR + (w * 16 + il) * 16);
            float4 r0 = Rp[0], r1 = Rp[1], r2 = Rp[2], r3 = Rp[3];
            const float4* d0 = (const float4*)(Db + (il * 32 + n) * 16);
            const float4* d1 = (const float4*)(Db + (il * 32 + n + 16) * 16);
            float rd0 = 0.f, rd1 = 0.f;
            {
                float4 a = d0[0]; rd0 += r0.x*a.x + r0.y*a.y + r0.z*a.z + r0.w*a.w;
                a = d0[1];        rd0 += r1.x*a.x + r1.y*a.y + r1.z*a.z + r1.w*a.w;
                a = d0[2];        rd0 += r2.x*a.x + r2.y*a.y + r2.z*a.z + r2.w*a.w;
                a = d0[3];        rd0 += r3.x*a.x + r3.y*a.y + r3.z*a.z + r3.w*a.w;
                a = d1[0];        rd1 += r0.x*a.x + r0.y*a.y + r0.z*a.z + r0.w*a.w;
                a = d1[1];        rd1 += r1.x*a.x + r1.y*a.y + r1.z*a.z + r1.w*a.w;
                a = d1[2];        rd1 += r2.x*a.x + r2.y*a.y + r2.z*a.z + r2.w*a.w;
                a = d1[3];        rd1 += r3.x*a.x + r3.y*a.y + r3.z*a.z + r3.w*a.w;
            }
            float mk0 = mask[(size_t)(i0 + il) * NSEQ + j0 + n];
            float mk1 = mask[(size_t)(i0 + il) * NSEQ + j0 + 16 + n];
            s[0][k] = scale * (s[0][k] + rd0) + mk0;
            s[1][k] = scale * (s[1][k] + rd1) + mk1;
        }

        // ---- online softmax update (row reductions across 16-lane halves) --
        float p0[8], p1[8], alpha[8];
#pragma unroll
        for (int k = 0; k < 8; ++k) {
            float t = fmaxf(s[0][k], s[1][k]);
#pragma unroll
            for (int off = 1; off < 16; off <<= 1)
                t = fmaxf(t, __shfl_xor(t, off, 32));
            float mnew = fmaxf(m_r[k], t);
            alpha[k] = __expf(m_r[k] - mnew);
            m_r[k] = mnew;
            p0[k] = __expf(s[0][k] - mnew);
            p1[k] = __expf(s[1][k] - mnew);
            float rs = p0[k] + p1[k];
#pragma unroll
            for (int off = 1; off < 16; off <<= 1)
                rs += __shfl_xor(rs, off, 32);
            l_r[k] = l_r[k] * alpha[k] + rs;
        }
#pragma unroll
        for (int dc = 0; dc < 4; ++dc)
#pragma unroll
            for (int k = 0; k < 8; ++k) acc[dc][k] *= alpha[k];

        // ---- relayout P (C layout -> A layout) through LDS ----
#pragma unroll
        for (int k = 0; k < 8; ++k) {
            ldsP[(w * 16 + k + 8 * hi) * 32 + n]      = f2bf(p0[k]);
            ldsP[(w * 16 + k + 8 * hi) * 32 + n + 16] = f2bf(p1[k]);
        }
        __syncthreads();
        FragBF pa;
        pa.u[0] = *(const uint4*)(ldsP + (w * 16 + n) * 32 + 8 * hi);
        pa.u[1] = *(const uint4*)(ldsP + (w * 16 + n) * 32 + 16 + 8 * hi);

        // ---- ctx += P @ V (V pre-transposed: contiguous j) ----
#pragma unroll
        for (int dc = 0; dc < 4; ++dc) {
            const unsigned short* vrow =
                Vt + ((size_t)h * DVd + dc * 16 + n) * NSEQ + j0 + 8 * hi;
            FragBF vb;
            vb.u[0] = *(const uint4*)(vrow);
            vb.u[1] = *(const uint4*)(vrow + 16);
            acc[dc] = __builtin_amdgcn_wmma_f32_16x16x32_bf16(false, pa.v, false,
                                                              vb.v, (short)0,
                                                              acc[dc], false, false);
        }
        buf ^= 1;
    }

    // ---- normalize and write ctx[i][h*DV + d] ----
#pragma unroll
    for (int k = 0; k < 8; ++k) {
        float inv = 1.0f / l_r[k];
#pragma unroll
        for (int dc = 0; dc < 4; ++dc)
            ctx[(size_t)(i0 + k + 8 * hi) * (NHh * DVd) + h * DVd + dc * 16 + n] =
                acc[dc][k] * inv;
    }
}

// ---------------------------------------------------------------------------
extern "C" void kernel_launch(void* const* d_in, const int* in_sizes, int n_in,
                              void* d_out, int out_size, void* d_ws, size_t ws_size,
                              hipStream_t stream) {
    (void)in_sizes; (void)n_in; (void)out_size; (void)ws_size;
    const float* H    = (const float*)d_in[0];
    const float* Dp   = (const float*)d_in[1];
    const float* mask = (const float*)d_in[2];
    const float* W    = (const float*)d_in[3];
    const float* Wout = (const float*)d_in[4];
    float* out = (float*)d_out;

    char* ws = (char*)d_ws;
    size_t off = 0;
    auto alloc = [&](size_t bytes) -> void* {
        void* p = ws + off;
        off += (bytes + 255) & ~(size_t)255;
        return p;
    };
    unsigned short* Hbf   = (unsigned short*)alloc((size_t)NSEQ * DIN * 2);
    unsigned short* Wt    = (unsigned short*)alloc((size_t)NPROJ * DIN * 2);
    unsigned short* Woutt = (unsigned short*)alloc((size_t)DMODEL * (NHh * DVd) * 2);
    unsigned short* Qb    = (unsigned short*)alloc((size_t)NHh * NSEQ * DKd * 2);
    unsigned short* Kb    = (unsigned short*)alloc((size_t)NHh * NSEQ * DKd * 2);
    unsigned short* Vt    = (unsigned short*)alloc((size_t)NHh * DVd * NSEQ * 2);
    float*          Rf    = (float*)alloc((size_t)NHh * NSEQ * DRd * 4);
    float*          H2    = (float*)alloc((size_t)NSEQ * NPROJ * 4);
    // H2 is dead after reformat; reuse its region for ctx / ctx_bf16
    float*          ctx   = H2;
    unsigned short* ctxbf = (unsigned short*)((char*)H2 +
                             (((size_t)NSEQ * NHh * DVd * 4 + 255) & ~(size_t)255));

    // 1) precision conversions / transposes
    cvt_bf16_kernel<<<1024, 256, 0, stream>>>(H, Hbf, NSEQ * DIN);
    cvt_tr_bf16_kernel<<<1024, 256, 0, stream>>>(W, Wt, DIN, NPROJ);
    cvt_tr_bf16_kernel<<<512, 256, 0, stream>>>(Wout, Woutt, NHh * DVd, DMODEL);

    // 2) projection GEMM: H2[N][NPROJ] = H @ W
    gemm_bf16_kernel<<<dim3(NPROJ / 16, NSEQ / 16), 32, 0, stream>>>(
        Hbf, Wt, H2, NSEQ, NPROJ, DIN);

    // 3) split into per-head Q / K / V^T / R
    reformat_kernel<<<NSEQ, 256, 0, stream>>>(H2, Qb, Kb, Vt, Rf);

    // 4) fused attention (TDM double-buffered D streaming; all heads/workgroup)
    attn_kernel<<<NSEQ / 16, 256, ATTN_LDS_BYTES, stream>>>(
        Qb, Kb, Vt, Rf, Dp, mask, ctx);

    // 5) output projection: out = ctx @ Wout
    cvt_bf16_kernel<<<1024, 256, 0, stream>>>(ctx, ctxbf, NSEQ * NHh * DVd);
    gemm_bf16_kernel<<<dim3(DMODEL / 16, NSEQ / 16), 32, 0, stream>>>(
        ctxbf, Woutt, out, NSEQ, DMODEL, NHh * DVd);
}
